// MoEFFN_49864570307106
// MI455X (gfx1250) — matrix operand order repaired and checked by reading, not compiled, not run on hardware
//
#include <hip/hip_runtime.h>
#include <hip/hip_bf16.h>

// ---------- problem constants ----------
#define NTOK 2048      // B*T
#define DDIM 1024
#define NEXP 32
#define HDIM 256
#define TOPK 4
#define NSH  2
#define HSH  1024

typedef __attribute__((ext_vector_type(16))) _Float16 v16h;
typedef __attribute__((ext_vector_type(8)))  _Float16 v8h;
typedef __attribute__((ext_vector_type(4)))  _Float16 v4h;
typedef __attribute__((ext_vector_type(8)))  float    v8f;
typedef __attribute__((ext_vector_type(4)))  int      v4i;

// ---------- CDNA5 async global->LDS path (guarded; sync fallback) ----------
#if defined(__has_builtin)
#if __has_builtin(__builtin_amdgcn_global_load_async_to_lds_b128)
#define USE_ASYNC_LDS 1
#endif
#endif
#ifndef USE_ASYNC_LDS
#define USE_ASYNC_LDS 0
#endif

#if USE_ASYNC_LDS
#define AS1 __attribute__((address_space(1)))
#define AS3 __attribute__((address_space(3)))
__device__ __forceinline__ void async_cp16(const _Float16* g, _Float16* l) {
    // 16B per lane, memory -> LDS, tracked by ASYNCcnt.
    AS1 v4i* gp = (AS1 v4i*)(AS1 void*)(void*)const_cast<_Float16*>(g);
    AS3 v4i* lp = (AS3 v4i*)(AS3 void*)(void*)l;
    __builtin_amdgcn_global_load_async_to_lds_b128(gp, lp, 0, 0);
}
__device__ __forceinline__ void wait_async0() {
#if __has_builtin(__builtin_amdgcn_s_wait_asynccnt)
    __builtin_amdgcn_s_wait_asynccnt(0);
#else
    asm volatile("s_wait_asynccnt 0" ::: "memory");
#endif
}
#endif

// ---------- wave32 reductions ----------
__device__ __forceinline__ float wave_max32(float v) {
    for (int m = 16; m > 0; m >>= 1) v = fmaxf(v, __shfl_xor(v, m, 32));
    return v;
}
__device__ __forceinline__ float wave_sum32(float v) {
    for (int m = 16; m > 0; m >>= 1) v += __shfl_xor(v, m, 32);
    return v;
}

// ---------- WMMA fragment gathers from LDS tiles (row stride = 64 halfs) ----
// A tile 16x64 f16: frag elem j -> row = lane&15, K = kb + j + 16*(lane>=16)
__device__ __forceinline__ v16h a_frag64(const _Float16* t, int kb, int lane) {
    int m = lane & 15, hi = lane >> 4;
    const _Float16* p = t + m * 64 + kb + 16 * hi;
    union { v16h v; v8h h[2]; } u;
    u.h[0] = *(const v8h*)p;
    u.h[1] = *(const v8h*)(p + 8);
    return u.v;
}
// B tile Fx64 f16 (B[k][n] = W[fcol+n][k]): frag elem j -> n = lane&15, K = kb + j + 16*(lane>=16)
__device__ __forceinline__ v16h b_frag64(const _Float16* t, int kb, int fcol, int lane) {
    int n = lane & 15, hi = lane >> 4;
    return *(const v16h*)(t + (size_t)(fcol + n) * 64 + kb + 16 * hi);
}

__device__ __forceinline__ float silu_f(float g) {
    return g / (1.0f + __expf(-g));
}

#define WMMA16(acc, a, b) \
    acc = __builtin_amdgcn_wmma_f32_16x16x32_f16(false, a, false, b, (short)0, acc, false, false)

// ---------- K0: zero counters ----------
__global__ void k_init(int* p) {
    if (threadIdx.x < 64) p[threadIdx.x] = 0;
}

// ---------- K1: x fp32 -> f16 ----------
__global__ void __launch_bounds__(256) k_cvt_x(const float* __restrict__ x,
                                               _Float16* __restrict__ xh) {
    size_t i = ((size_t)blockIdx.x * blockDim.x + threadIdx.x) * 4;
    float4 v = *(const float4*)(x + i);
    v4h o = { (_Float16)v.x, (_Float16)v.y, (_Float16)v.z, (_Float16)v.w };
    *(v4h*)(xh + i) = o;
}

// ---------- K2: router (one wave per token, lane == expert) ----------
__global__ void __launch_bounds__(256) k_router(const float* __restrict__ x,
                                                const float* __restrict__ Wr,
                                                const float* __restrict__ rbias,
                                                int* __restrict__ cnt,
                                                int* __restrict__ topk_idx,
                                                float* __restrict__ topk_w) {
    int wave = threadIdx.x >> 5;
    int lane = threadIdx.x & 31;
    int token = blockIdx.x * 8 + wave;

    const float4* xr = (const float4*)(x + (size_t)token * DDIM);
    const float4* wr = (const float4*)(Wr + (size_t)lane * DDIM);
    float acc = 0.f;
    for (int i = 0; i < DDIM / 4; ++i) {
        float4 a = xr[i], b = wr[i];
        acc += a.x * b.x + a.y * b.y + a.z * b.z + a.w * b.w;
    }
    float mx = wave_max32(acc);
    float p  = __expf(acc - mx);
    float sm = wave_sum32(p);
    float score = p / sm;
    float bl = acc + rbias[lane];
    int   sel[TOPK];
    float sw[TOPK];
    float tsum = 0.f;
    for (int k = 0; k < TOPK; ++k) {
        float mv = wave_max32(bl);
        unsigned long long msk = __ballot(bl == mv);
        int s = __ffsll(msk) - 1;           // lowest index on ties (matches top_k)
        sel[k] = s;
        sw[k]  = __shfl(score, s, 32);
        tsum  += sw[k];
        if (lane == s) bl = -3.4e38f;
    }
    if (lane < TOPK) {
        int   myi = sel[lane];
        float myw = sw[lane] / tsum;
        topk_idx[token * TOPK + lane] = myi;
        topk_w[token * TOPK + lane]   = myw;
        atomicAdd(&cnt[myi], 1);
    }
}

// ---------- K3: exclusive prefix over 32 counts ----------
__global__ void k_prefix(const int* __restrict__ cnt, int* __restrict__ off) {
    if (threadIdx.x == 0) {
        int a = 0;
        for (int e = 0; e < NEXP; ++e) { off[e] = a; a += cnt[e]; }
    }
}

// ---------- K4: fill compact expert-major slot lists ----------
__global__ void __launch_bounds__(256) k_fill(const int* __restrict__ topk_idx,
                                              const float* __restrict__ topk_w,
                                              const int* __restrict__ off,
                                              int* __restrict__ fill,
                                              int* __restrict__ slot_token,
                                              float* __restrict__ slot_w) {
    int a = blockIdx.x * blockDim.x + threadIdx.x;
    if (a >= NTOK * TOPK) return;
    int e   = topk_idx[a];
    int pos = atomicAdd(&fill[e], 1);
    int s   = off[e] + pos;
    slot_token[s] = a >> 2;
    slot_w[s]     = topk_w[a];
}

// ---------- K5: routed gate/up GEMM + SiLU*u*cw -> hbuf (f16) ----------
// grid (E, H/64); block 256 = 8 waves: waves 0-3 -> gate, 4-7 -> up.
__global__ void __launch_bounds__(256) k_moe_gu(const _Float16* __restrict__ xh,
                                                const float* __restrict__ Wg,
                                                const float* __restrict__ Wu,
                                                const int* __restrict__ cnt,
                                                const int* __restrict__ off,
                                                const int* __restrict__ slot_token,
                                                const float* __restrict__ slot_w,
                                                _Float16* __restrict__ hbuf) {
    __shared__ __attribute__((aligned(32))) _Float16 xs[16 * 64];
    __shared__ __attribute__((aligned(32))) _Float16 ws0[64 * 64];
    __shared__ __attribute__((aligned(32))) _Float16 ws1[64 * 64];
    __shared__ float gx[16 * 64];
    __shared__ float cwS[16];

    int e     = blockIdx.x;
    int hbase = blockIdx.y * 64;
    int tid   = threadIdx.x, lane = tid & 31, wave = tid >> 5;
    int n     = lane & 15, hi = lane >> 4;
    int fcol  = (wave & 3) * 16;
    int count = cnt[e], base = off[e];
    const _Float16* myW = (wave < 4) ? ws0 : ws1;

    for (int t0 = 0; t0 < count; t0 += 16) {
        int rows = min(16, count - t0);
        __syncthreads();
        if (tid < 16)
            cwS[tid] = (tid < rows) ? slot_w[base + t0 + tid] : 0.f;
        __syncthreads();

        // Hoist loop-invariant A-tile copy addresses (wave 0 issues copies).
        // Invalid rows are clamped to a valid token: garbage only lands in
        // A-rows whose output rows are never stored.
#if USE_ASYNC_LDS
        const _Float16* gsrc[4];
        _Float16*       ldst[4];
        if (wave == 0) {
            for (int q = 0; q < 4; ++q) {
                int idx = q * 32 + lane;
                int r = idx >> 3, cb = (idx & 7) * 8;  // 8 halfs = 16B
                int rr  = (r < rows) ? r : 0;
                int tok = slot_token[base + t0 + rr];
                gsrc[q] = xh + (size_t)tok * DDIM + cb;
                ldst[q] = xs + r * 64 + cb;
            }
        }
#endif
        v8f acc = {};
        for (int k0 = 0; k0 < DDIM; k0 += 64) {
#if USE_ASYNC_LDS
            if (wave == 0) {
                for (int q = 0; q < 4; ++q) async_cp16(gsrc[q] + k0, ldst[q]);
            }
#else
            {
                int r = tid >> 4, c = (tid & 15) * 4;
                int rr  = (r < rows) ? r : 0;
                int tok = slot_token[base + t0 + rr];
                uint2 v = *(const uint2*)(xh + (size_t)tok * DDIM + k0 + c);
                *(uint2*)(xs + r * 64 + c) = v;
            }
#endif
            // ---- weight tiles 64x64, f32 -> f16 ----
            {
                int f = tid >> 2, c0 = (tid & 3) * 16;
                const float* g0 = Wg + ((size_t)e * HDIM + hbase + f) * DDIM + k0 + c0;
                const float* g1 = Wu + ((size_t)e * HDIM + hbase + f) * DDIM + k0 + c0;
                for (int h8 = 0; h8 < 2; ++h8) {
                    v8h o0, o1;
                    for (int j = 0; j < 8; ++j) {
                        o0[j] = (_Float16)g0[h8 * 8 + j];
                        o1[j] = (_Float16)g1[h8 * 8 + j];
                    }
                    *(v8h*)(ws0 + f * 64 + c0 + h8 * 8) = o0;
                    *(v8h*)(ws1 + f * 64 + c0 + h8 * 8) = o1;
                }
                __builtin_prefetch(g0 + 64, 0, 0);
            }
#if USE_ASYNC_LDS
            if (wave == 0) wait_async0();
#endif
            __syncthreads();
            v16h a0 = a_frag64(xs, 0, lane);
            v16h b0 = b_frag64(myW, 0, fcol, lane);
            WMMA16(acc, a0, b0);
            v16h a1 = a_frag64(xs, 32, lane);
            v16h b1 = b_frag64(myW, 32, fcol, lane);
            WMMA16(acc, a1, b1);
            __syncthreads();
        }
        // epilogue: gate waves publish g, up waves fuse silu(g)*u*cw
        if (wave < 4) {
            for (int r = 0; r < 8; ++r) {
                int M = r + 8 * hi;
                gx[M * 64 + fcol + n] = acc[r];
            }
        }
        __syncthreads();
        if (wave >= 4) {
            for (int r = 0; r < 8; ++r) {
                int M = r + 8 * hi;
                if (M < rows) {
                    float g  = gx[M * 64 + fcol + n];
                    float hv = silu_f(g) * acc[r] * cwS[M];
                    hbuf[(size_t)(base + t0 + M) * HDIM + hbase + fcol + n] = (_Float16)hv;
                }
            }
        }
    }
}

// ---------- K6: routed down GEMM, atomic-add into out ----------
// grid (E, D/128); block 256 = 8 waves, tile 16 slots x 128 d.
__global__ void __launch_bounds__(256) k_moe_down(const _Float16* __restrict__ hbuf,
                                                  const float* __restrict__ Wd,
                                                  const int* __restrict__ cnt,
                                                  const int* __restrict__ off,
                                                  const int* __restrict__ slot_token,
                                                  float* __restrict__ out) {
    __shared__ __attribute__((aligned(32))) _Float16 hsA[16 * 64];
    __shared__ __attribute__((aligned(32))) _Float16 wds[128 * 64];
    __shared__ int tokS[16];

    int e     = blockIdx.x;
    int dbase = blockIdx.y * 128;
    int tid   = threadIdx.x, lane = tid & 31, wave = tid >> 5;
    int n     = lane & 15, hi = lane >> 4;
    int fcol  = wave * 16;
    int count = cnt[e], base = off[e];

    for (int t0 = 0; t0 < count; t0 += 16) {
        int rows = min(16, count - t0);
        __syncthreads();
        if (tid < 16) tokS[tid] = (tid < rows) ? slot_token[base + t0 + tid] : -1;
        __syncthreads();

#if USE_ASYNC_LDS
        const _Float16* gsrc[4];
        _Float16*       ldst[4];
        if (wave == 0) {
            for (int q = 0; q < 4; ++q) {
                int idx = q * 32 + lane;
                int r = idx >> 3, cb = (idx & 7) * 8;
                int rr = (r < rows) ? r : 0;
                gsrc[q] = hbuf + (size_t)(base + t0 + rr) * HDIM + cb;
                ldst[q] = hsA + r * 64 + cb;
            }
        }
#endif
        v8f acc = {};
        for (int k0 = 0; k0 < HDIM; k0 += 64) {
#if USE_ASYNC_LDS
            if (wave == 0) {
                for (int q = 0; q < 4; ++q) async_cp16(gsrc[q] + k0, ldst[q]);
            }
#else
            {
                int r = tid >> 4, c = (tid & 15) * 4;
                int rr = (r < rows) ? r : 0;
                uint2 v = *(const uint2*)(hbuf + (size_t)(base + t0 + rr) * HDIM + k0 + c);
                *(uint2*)(hsA + r * 64 + c) = v;
            }
#endif
            // ---- weight tile 128x64, f32 -> f16 ----
            {
                int f = tid >> 1, c0 = (tid & 1) * 32;
                const float* src = Wd + ((size_t)e * DDIM + dbase + f) * HDIM + k0 + c0;
                for (int h8 = 0; h8 < 4; ++h8) {
                    v8h o;
                    for (int j = 0; j < 8; ++j) o[j] = (_Float16)src[h8 * 8 + j];
                    *(v8h*)(wds + f * 64 + c0 + h8 * 8) = o;
                }
                __builtin_prefetch(src + 64, 0, 0);
            }
#if USE_ASYNC_LDS
            if (wave == 0) wait_async0();
#endif
            __syncthreads();
            v16h a0 = a_frag64(hsA, 0, lane);
            v16h b0 = b_frag64(wds, 0, fcol, lane);
            WMMA16(acc, a0, b0);
            v16h a1 = a_frag64(hsA, 32, lane);
            v16h b1 = b_frag64(wds, 32, fcol, lane);
            WMMA16(acc, a1, b1);
            __syncthreads();
        }
        for (int r = 0; r < 8; ++r) {
            int M = r + 8 * hi;
            if (M < rows) {
                int tok = tokS[M];
                atomicAdd(out + (size_t)tok * DDIM + dbase + fcol + n, acc[r]);
            }
        }
    }
}

// ---------- K7: shared-expert gate/up GEMM -> hsbuf (f16) ----------
// grid (N/16, HS/64, S); block 256: waves 0-3 gate(Sg), 4-7 up(Su).
__global__ void __launch_bounds__(256) k_sh_gu(const _Float16* __restrict__ xh,
                                               const float* __restrict__ Sg,
                                               const float* __restrict__ Su,
                                               _Float16* __restrict__ hsbuf) {
    __shared__ __attribute__((aligned(32))) _Float16 xs[16 * 64];
    __shared__ __attribute__((aligned(32))) _Float16 ws0[64 * 64];
    __shared__ __attribute__((aligned(32))) _Float16 ws1[64 * 64];
    __shared__ float gx[16 * 64];

    int tbase = blockIdx.x * 16;
    int hbase = blockIdx.y * 64;
    int s     = blockIdx.z;
    int tid   = threadIdx.x, lane = tid & 31, wave = tid >> 5;
    int n     = lane & 15, hi = lane >> 4;
    int fcol  = (wave & 3) * 16;
    const _Float16* myW = (wave < 4) ? ws0 : ws1;

#if USE_ASYNC_LDS
    const _Float16* gsrc[4];
    _Float16*       ldst[4];
    if (wave == 0) {
        for (int q = 0; q < 4; ++q) {
            int idx = q * 32 + lane;
            int r = idx >> 3, cb = (idx & 7) * 8;
            gsrc[q] = xh + (size_t)(tbase + r) * DDIM + cb;
            ldst[q] = xs + r * 64 + cb;
        }
    }
#endif
    v8f acc = {};
    for (int k0 = 0; k0 < DDIM; k0 += 64) {
#if USE_ASYNC_LDS
        if (wave == 0) {
            for (int q = 0; q < 4; ++q) async_cp16(gsrc[q] + k0, ldst[q]);
        }
#else
        {
            int r = tid >> 4, c = (tid & 15) * 4;
            uint2 v = *(const uint2*)(xh + (size_t)(tbase + r) * DDIM + k0 + c);
            *(uint2*)(xs + r * 64 + c) = v;
        }
#endif
        {
            int f = tid >> 2, c0 = (tid & 3) * 16;
            const float* g0 = Sg + ((size_t)s * HSH + hbase + f) * DDIM + k0 + c0;
            const float* g1 = Su + ((size_t)s * HSH + hbase + f) * DDIM + k0 + c0;
            for (int h8 = 0; h8 < 2; ++h8) {
                v8h o0, o1;
                for (int j = 0; j < 8; ++j) {
                    o0[j] = (_Float16)g0[h8 * 8 + j];
                    o1[j] = (_Float16)g1[h8 * 8 + j];
                }
                *(v8h*)(ws0 + f * 64 + c0 + h8 * 8) = o0;
                *(v8h*)(ws1 + f * 64 + c0 + h8 * 8) = o1;
            }
            __builtin_prefetch(g0 + 64, 0, 0);
        }
#if USE_ASYNC_LDS
        if (wave == 0) wait_async0();
#endif
        __syncthreads();
        v16h a0 = a_frag64(xs, 0, lane);
        v16h b0 = b_frag64(myW, 0, fcol, lane);
        WMMA16(acc, a0, b0);
        v16h a1 = a_frag64(xs, 32, lane);
        v16h b1 = b_frag64(myW, 32, fcol, lane);
        WMMA16(acc, a1, b1);
        __syncthreads();
    }
    if (wave < 4) {
        for (int r = 0; r < 8; ++r) {
            int M = r + 8 * hi;
            gx[M * 64 + fcol + n] = acc[r];
        }
    }
    __syncthreads();
    if (wave >= 4) {
        for (int r = 0; r < 8; ++r) {
            int M = r + 8 * hi;
            float g  = gx[M * 64 + fcol + n];
            float hv = silu_f(g) * acc[r];
            hsbuf[((size_t)s * NTOK + tbase + M) * HSH + hbase + fcol + n] = (_Float16)hv;
        }
    }
}

// ---------- K8: shared-expert down GEMM (plain store; runs before K6) ----------
// grid (N/16, D/128); accumulates over both shared experts in-register.
__global__ void __launch_bounds__(256) k_sh_down(const _Float16* __restrict__ hsbuf,
                                                 const float* __restrict__ Sd,
                                                 float* __restrict__ out) {
    __shared__ __attribute__((aligned(32))) _Float16 hsA[16 * 64];
    __shared__ __attribute__((aligned(32))) _Float16 wds[128 * 64];

    int tbase = blockIdx.x * 16;
    int dbase = blockIdx.y * 128;
    int tid   = threadIdx.x, lane = tid & 31, wave = tid >> 5;
    int n     = lane & 15, hi = lane >> 4;
    int fcol  = wave * 16;

#if USE_ASYNC_LDS
    int aq_r[4], aq_cb[4];
    _Float16* ldst[4];
    if (wave == 0) {
        for (int q = 0; q < 4; ++q) {
            int idx = q * 32 + lane;
            aq_r[q]  = idx >> 3;
            aq_cb[q] = (idx & 7) * 8;
            ldst[q]  = hsA + aq_r[q] * 64 + aq_cb[q];
        }
    }
#endif
    v8f acc = {};
    for (int s = 0; s < NSH; ++s) {
#if USE_ASYNC_LDS
        const _Float16* gsrc[4];
        if (wave == 0) {
            for (int q = 0; q < 4; ++q)
                gsrc[q] = hsbuf + ((size_t)s * NTOK + tbase + aq_r[q]) * HSH + aq_cb[q];
        }
#endif
        for (int k0 = 0; k0 < HSH; k0 += 64) {
#if USE_ASYNC_LDS
            if (wave == 0) {
                for (int q = 0; q < 4; ++q) async_cp16(gsrc[q] + k0, ldst[q]);
            }
#else
            {
                int r = tid >> 4, c = (tid & 15) * 4;
                uint2 v = *(const uint2*)(hsbuf + ((size_t)s * NTOK + tbase + r) * HSH + k0 + c);
                *(uint2*)(hsA + r * 64 + c) = v;
            }
#endif
            {
                int f = tid >> 1, c0 = (tid & 1) * 32;
                const float* src = Sd + ((size_t)s * DDIM + dbase + f) * HSH + k0 + c0;
                for (int h8 = 0; h8 < 4; ++h8) {
                    v8h o;
                    for (int j = 0; j < 8; ++j) o[j] = (_Float16)src[h8 * 8 + j];
                    *(v8h*)(wds + f * 64 + c0 + h8 * 8) = o;
                }
                __builtin_prefetch(src + 64, 0, 0);
            }
#if USE_ASYNC_LDS
            if (wave == 0) wait_async0();
#endif
            __syncthreads();
            v16h a0 = a_frag64(hsA, 0, lane);
            v16h b0 = b_frag64(wds, 0, fcol, lane);
            WMMA16(acc, a0, b0);
            v16h a1 = a_frag64(hsA, 32, lane);
            v16h b1 = b_frag64(wds, 32, fcol, lane);
            WMMA16(acc, a1, b1);
            __syncthreads();
        }
    }
    for (int r = 0; r < 8; ++r) {
        int M = r + 8 * hi;
        out[(size_t)(tbase + M) * DDIM + dbase + fcol + n] = acc[r];
    }
}

// ---------- host launcher ----------
extern "C" void kernel_launch(void* const* d_in, const int* in_sizes, int n_in,
                              void* d_out, int out_size, void* d_ws, size_t ws_size,
                              hipStream_t stream) {
    (void)in_sizes; (void)n_in; (void)out_size; (void)ws_size;
    const float* x    = (const float*)d_in[0];
    const float* Wr   = (const float*)d_in[1];
    const float* rb   = (const float*)d_in[2];
    const float* Wg   = (const float*)d_in[3];
    const float* Wu   = (const float*)d_in[4];
    const float* Wd   = (const float*)d_in[5];
    const float* Sg   = (const float*)d_in[6];
    const float* Su   = (const float*)d_in[7];
    const float* Sd   = (const float*)d_in[8];
    float*       out  = (float*)d_out;

    char* w = (char*)d_ws;
    int*   cnt        = (int*)w;                 // [32]
    int*   fil        = cnt + 32;                // [32]
    int*   off        = cnt + 64;                // [32]
    int*   topk_idx   = (int*)(w + 512);                   // [2048*4]
    float* topk_w     = (float*)(w + 512 + 32768);         // [2048*4]
    int*   slot_token = (int*)(w + 512 + 65536);           // [8192]
    float* slot_w     = (float*)(w + 512 + 98304);         // [8192]
    _Float16* xh      = (_Float16*)(w + 131584);           // [2048*1024]
    _Float16* hbuf    = xh + (size_t)NTOK * DDIM;          // [8192*256]
    _Float16* hsbuf   = hbuf + (size_t)NTOK * TOPK * HDIM; // [2*2048*1024]

    k_init<<<1, 64, 0, stream>>>(cnt);
    k_cvt_x<<<(NTOK * DDIM) / (256 * 4), 256, 0, stream>>>(x, xh);
    k_router<<<NTOK / 8, 256, 0, stream>>>(x, Wr, rb, cnt, topk_idx, topk_w);
    k_prefix<<<1, 32, 0, stream>>>(cnt, off);
    k_fill<<<(NTOK * TOPK) / 256, 256, 0, stream>>>(topk_idx, topk_w, off, fil,
                                                    slot_token, slot_w);
    k_sh_gu<<<dim3(NTOK / 16, HSH / 64, NSH), 256, 0, stream>>>(xh, Sg, Su, hsbuf);
    k_moe_gu<<<dim3(NEXP, HDIM / 64), 256, 0, stream>>>(xh, Wg, Wu, cnt, off,
                                                        slot_token, slot_w, hbuf);
    k_sh_down<<<dim3(NTOK / 16, DDIM / 128), 256, 0, stream>>>(hsbuf, Sd, out);
    k_moe_down<<<dim3(NEXP, DDIM / 128), 256, 0, stream>>>(hbuf, Wd, cnt, off,
                                                           slot_token, out);
}
